// TopoGAT_70239895159064
// MI455X (gfx1250) — matrix operand order repaired and checked by reading, not compiled, not run on hardware
//
#include <hip/hip_runtime.h>
#include <cstdint>

typedef __attribute__((ext_vector_type(16))) _Float16 v16h;
typedef __attribute__((ext_vector_type(8)))  _Float16 v8h;
typedef __attribute__((ext_vector_type(8)))  float    v8f;

#define GN 4096      // nodes
#define GH 8         // heads
#define GO 64        // per-head output dim
#define NEG_SLOPE 0.2f

// K index inside a 16x32 f16 A-fragment: VGPR pair v (0..7), lane half ksel.
__device__ __forceinline__ int amapk(int v, int ksel) {
    return (v << 1) + (ksel << 4);
}

// Swizzled Wh layout: f16 records of 16 elements.
//   rec(h, jt, q, l) = Whb[ ((((h*(GN/32) + jt)*4 + q)*32) + l) * 16 ]
// Lane l (<16): column n=l, K=0..15 of 32-row tile jt; lane l>=16: column l-16, K=16..31.
// A v16h load of one record pair IS the WMMA B fragment for (32xK, 16-col block q).

// ---------------------------------------------------------------------------
// Wh[h] = X @ W[h]  (f32 accumulate via v_wmma_f32_16x16x32_f16),
// stored pre-swizzled in B-fragment layout; also computes s,d score vectors.
// One wave per 16x16 output tile; 8 waves (2 row blocks x 4 col blocks, same
// head) per 256-thread block.
// ---------------------------------------------------------------------------
__global__ __launch_bounds__(256) void gat_transform(
    const float* __restrict__ X, const float* __restrict__ W,
    const float* __restrict__ asrc, const float* __restrict__ adst,
    _Float16* __restrict__ Whb, float* __restrict__ sbuf,
    float* __restrict__ dbuf, int F)
{
    __shared__ float ldsA[8][16 * 32];   // per-wave A tile (16 rows x 32 K), f32
    __shared__ float ldsB[8][32 * 16];   // per-wave B tile (32 K x 16 cols), f32
    __shared__ float ldsS[8][16];
    __shared__ float ldsD[8][16];

    const int lane  = threadIdx.x & 31;
    const int wslot = threadIdx.x >> 5;
    const int wave  = blockIdx.x * 8 + wslot;
    const int h  = wave >> 10;            // 1024 waves per head (256 rb * 4 cb)
    const int rb = (wave >> 2) & 255;
    const int cb = wave & 3;
    const int rowBase = rb * 16;
    const int colBase = cb * 16;
    const int nidx = lane & 15;           // row (A) / column (B,C)
    const int ksel = lane >> 4;

    v8f acc = {0.f, 0.f, 0.f, 0.f, 0.f, 0.f, 0.f, 0.f};

    for (int k0 = 0; k0 < F; k0 += 32) {
        // Stage A tile: X[rowBase..+16][k0..+32]  (coalesced float4 loads)
        #pragma unroll
        for (int it = 0; it < 4; ++it) {
            int slot = lane + 32 * it;            // 128 float4 slots
            int r = slot >> 3, c4 = slot & 7;
            const float4 va = *(const float4*)&X[(size_t)(rowBase + r) * F + k0 + c4 * 4];
            *(float4*)&ldsA[wslot][r * 32 + c4 * 4] = va;
        }
        // Stage B tile: W[h][k0..+32][colBase..+16]
        #pragma unroll
        for (int it = 0; it < 4; ++it) {
            int slot = lane + 32 * it;
            int r = slot >> 2, c4 = slot & 3;
            const float4 vb = *(const float4*)&W[((size_t)h * F + k0 + r) * GO + colBase + c4 * 4];
            *(float4*)&ldsB[wslot][r * 16 + c4 * 4] = vb;
        }
        __syncthreads();

        v16h a, b;
        #pragma unroll
        for (int v = 0; v < 8; ++v) {             // A-fragment gather (contiguous k = 16*ksel + s)
            int kk = amapk(v, ksel);
            a[2 * v]     = (_Float16)ldsA[wslot][nidx * 32 + kk];
            a[2 * v + 1] = (_Float16)ldsA[wslot][nidx * 32 + kk + 1];
        }
        const int kb = ksel << 4;
        #pragma unroll
        for (int v = 0; v < 8; ++v) {             // B-fragment gather (contiguous k = 16*ksel + s)
            b[2 * v]     = (_Float16)ldsB[wslot][(kb + 2 * v) * 16 + nidx];
            b[2 * v + 1] = (_Float16)ldsB[wslot][(kb + 2 * v + 1) * 16 + nidx];
        }
        acc = __builtin_amdgcn_wmma_f32_16x16x32_f16(false, a, false, b,
                                                     (short)0, acc, false, false);
        __syncthreads();
    }

    // ---- Store C tile into swizzled B-fragment layout (one b128 per lane) ----
    {
        const int jt   = rowBase >> 5;            // 32-row tile index
        const int half = (rowBase >> 4) & 1;      // which K-half of the tile
        v8h pk;
        #pragma unroll
        for (int v = 0; v < 8; ++v) pk[v] = (_Float16)acc[v];
        // record (16*half + nidx), K positions m = v + 8*ksel -> byte half ksel
        _Float16* dst = &Whb[(((((size_t)h * (GN / 32) + jt) * 4 + cb) * 32)
                              + 16 * half + nidx) * 16 + ksel * 8];
        *(v8h*)dst = pk;
    }

    // ---- Fused s,d:  s[i] = Wh[i,:].a_src,  d[i] = Wh[i,:].a_dst ----
    {
        const float aS = asrc[h * GO + colBase + nidx];
        const float aD = adst[h * GO + colBase + nidx];
        float sp[8], dp[8];
        #pragma unroll
        for (int v = 0; v < 8; ++v) { sp[v] = acc[v] * aS; dp[v] = acc[v] * aD; }
        #pragma unroll
        for (int v = 0; v < 8; ++v) {             // reduce over 16 cols (half-wave)
            #pragma unroll
            for (int off = 1; off < 16; off <<= 1) {
                sp[v] += __shfl_xor(sp[v], off, 32);
                dp[v] += __shfl_xor(dp[v], off, 32);
            }
        }
        if (nidx == 0) {
            #pragma unroll
            for (int v = 0; v < 8; ++v) {
                ldsS[wslot][v + (ksel << 3)] = sp[v];
                ldsD[wslot][v + (ksel << 3)] = dp[v];
            }
        }
        __syncthreads();
        if (threadIdx.x < 64) {                   // combine 4 col-block waves
            int grp = threadIdx.x >> 5;           // row block within this block
            int isD = (threadIdx.x >> 4) & 1;
            int row = threadIdx.x & 15;
            const float* src = isD ? &ldsD[0][0] : &ldsS[0][0];
            float t = 0.f;
            #pragma unroll
            for (int q = 0; q < 4; ++q) t += src[(grp * 4 + q) * 16 + row];
            float* dst = isD ? dbuf : sbuf;
            dst[h * GN + (blockIdx.x * 2 + grp) * 16 + row] = t;
        }
    }
}

// ---------------------------------------------------------------------------
// Fused masked softmax + aggregation + ELU (flash-attention style).
// Block = 8 waves; wave w handles head w for one 16-row block.
// ---------------------------------------------------------------------------
__global__ __launch_bounds__(256) void gat_attn(
    const float* __restrict__ adj, const _Float16* __restrict__ Whb,
    const float* __restrict__ s, const float* __restrict__ d,
    float* __restrict__ out)
{
    __shared__ __align__(16) float ldsP[GH][16 * 34];  // stride 34: f64-aligned, conflict-free

    const int lane = threadIdx.x & 31;
    const int h    = threadIdx.x >> 5;    // wave == head
    const int i0   = blockIdx.x * 16;
    const int mrow = lane & 15;
    const int ksel = lane >> 4;

    float sreg[16];
    #pragma unroll
    for (int r = 0; r < 16; ++r) sreg[r] = s[h * GN + i0 + r];

    // ---- Pass 1: per-row max of masked leaky_relu(s_i + d_j) ----
    float mx[16];
    #pragma unroll
    for (int r = 0; r < 16; ++r) mx[r] = -1e30f;
    for (int j0 = 0; j0 < GN; j0 += 32) {
        float dv = d[h * GN + j0 + lane];
        #pragma unroll
        for (int r = 0; r < 16; ++r) {
            float av = adj[(size_t)(i0 + r) * GN + j0 + lane];
            float e = sreg[r] + dv;
            e = e > 0.f ? e : NEG_SLOPE * e;
            mx[r] = (av > 0.f && e > mx[r]) ? e : mx[r];
        }
    }
    #pragma unroll
    for (int r = 0; r < 16; ++r) {
        float m = mx[r];
        #pragma unroll
        for (int off = 16; off > 0; off >>= 1) m = fmaxf(m, __shfl_xor(m, off, 32));
        mx[r] = m;
    }

    // ---- Pass 2: P = exp(e - m) (masked), out += P @ Wh via WMMA ----
    v8f c[4];
    #pragma unroll
    for (int q = 0; q < 4; ++q) { v8f z = {0, 0, 0, 0, 0, 0, 0, 0}; c[q] = z; }
    float rs[16];
    #pragma unroll
    for (int r = 0; r < 16; ++r) rs[r] = 0.f;

    for (int j0 = 0; j0 < GN; j0 += 32) {
        if (j0 + 32 < GN)                                 // global_prefetch_b8
            __builtin_prefetch(&adj[(size_t)i0 * GN + j0 + 32 + lane], 0, 0);
        float dv = d[h * GN + j0 + lane];
        #pragma unroll
        for (int r = 0; r < 16; ++r) {
            float av = adj[(size_t)(i0 + r) * GN + j0 + lane];
            float e = sreg[r] + dv;
            e = e > 0.f ? e : NEG_SLOPE * e;
            float p = (av > 0.f) ? __expf(e - mx[r]) : 0.f;
            rs[r] += p;
            ldsP[h][r * 34 + lane] = p;                   // lane = column j
        }
        __syncthreads();
        v16h a;
        #pragma unroll
        for (int v = 0; v < 8; ++v) {                     // transpose into A layout
            int kk = amapk(v, ksel);                      // kk is even
            float2 pr = *(const float2*)&ldsP[h][mrow * 34 + kk];
            a[2 * v]     = (_Float16)pr.x;
            a[2 * v + 1] = (_Float16)pr.y;
        }
        __syncthreads();
        const size_t tbase = (((size_t)h * (GN / 32) + (j0 >> 5)) * 4);
        #pragma unroll
        for (int q = 0; q < 4; ++q) {                     // 4 x 16 output columns
            // one 32-byte vector load per lane IS the B fragment
            const v16h b = *(const v16h*)&Whb[((tbase + q) * 32 + lane) * 16];
            c[q] = __builtin_amdgcn_wmma_f32_16x16x32_f16(false, a, false, b,
                                                          (short)0, c[q], false, false);
        }
    }

    // ---- row-sum reduce, normalize, ELU, interleaved store ----
    #pragma unroll
    for (int r = 0; r < 16; ++r) {
        float t = rs[r];
        #pragma unroll
        for (int off = 16; off > 0; off >>= 1) t += __shfl_xor(t, off, 32);
        rs[r] = 1.f / t;                  // self-loop guarantees t > 0
    }
    #pragma unroll
    for (int q = 0; q < 4; ++q) {
        #pragma unroll
        for (int v = 0; v < 8; ++v) {
            float inv = ksel ? rs[v + 8] : rs[v];
            float val = c[q][v] * inv;
            val = val > 0.f ? val : (__expf(val) - 1.f);   // ELU(alpha=1)
            int m = v + (ksel << 3);
            out[(size_t)(i0 + m) * (GH * GO) + h * GO + q * 16 + mrow] = val;
        }
    }
}

// ---------------------------------------------------------------------------
extern "C" void kernel_launch(void* const* d_in, const int* in_sizes, int n_in,
                              void* d_out, int out_size, void* d_ws, size_t ws_size,
                              hipStream_t stream)
{
    (void)in_sizes; (void)n_in; (void)out_size; (void)ws_size;
    const float* feat = (const float*)d_in[0];
    const float* adj  = (const float*)d_in[1];
    const float* W[3]  = {(const float*)d_in[2], (const float*)d_in[5], (const float*)d_in[8]};
    const float* As[3] = {(const float*)d_in[3], (const float*)d_in[6], (const float*)d_in[9]};
    const float* Ad[3] = {(const float*)d_in[4], (const float*)d_in[7], (const float*)d_in[10]};
    float* out = (float*)d_out;

    // Workspace layout (~12.3 MiB total)
    char* ws = (char*)d_ws;
    _Float16* Whb = (_Float16*)ws;                          // swizzled Wh, 4 MiB
    float* sbuf = (float*)(ws + (size_t)4 * 1024 * 1024);   // 128 KiB
    float* dbuf = sbuf + GH * GN;                           // 128 KiB
    float* h1   = dbuf + GH * GN;                           // 4096*512 f32 = 8 MiB

    const int Fdims[3]  = {256, GH * GO, GH * GO};
    const float* Xs[3]  = {feat, h1, out};   // layer3 reads d_out (written by layer2)
    float* Ys[3]        = {h1, out, out};    // safe: transform consumes X before attn writes Y

    for (int l = 0; l < 3; ++l) {
        const int tiles = GH * (GN / 16) * (GO / 16);       // 8192 waves, 8 per block
        gat_transform<<<tiles / 8, 256, 0, stream>>>(Xs[l], W[l], As[l], Ad[l],
                                                     Whb, sbuf, dbuf, Fdims[l]);
        gat_attn<<<GN / 16, 256, 0, stream>>>(adj, Whb, sbuf, dbuf, Ys[l]);
    }
}
